// Model_30313879175692
// MI455X (gfx1250) — matrix operand order repaired and checked by reading, not compiled, not run on hardware
//
#include <hip/hip_runtime.h>
#include <math.h>

// ---------------- model constants ----------------
#define BSZ   8
#define MT    48
#define ME    384
#define NP    10          // players
#define DSI   64
#define DSG   32
#define DD    256
#define DHH   1024
#define NHEAD 8
#define DK    32
#define S_TOK (BSZ*MT*NP) // 3840
#define E_TOK (BSZ*ME)    // 3072
#define G_TOK (BSZ*MT)    // 384

typedef __attribute__((ext_vector_type(16))) _Float16 v16h;
typedef __attribute__((ext_vector_type(8)))  float    v8f;

__device__ __forceinline__ void wait_asynccnt0()
{
#if __has_builtin(__builtin_amdgcn_s_wait_asynccnt)
    __builtin_amdgcn_s_wait_asynccnt(0);
#else
    asm volatile("s_wait_asynccnt 0x0" ::: "memory");
#endif
}

// async copy 16B global -> LDS (ASYNCcnt tracked, no VGPR round trip)
__device__ __forceinline__ void async_g2l_b128(unsigned lds_addr, const float* gaddr)
{
    asm volatile("global_load_async_to_lds_b128 %0, %1, off"
                 :: "v"(lds_addr), "v"(gaddr) : "memory");
}

__device__ __forceinline__ v8f wmma_f16(const v16h& a, const v16h& b, const v8f& c)
{
    return __builtin_amdgcn_wmma_f32_16x16x32_f16(false, a, false, b, (short)0, c, false, false);
}

// =======================================================================
// WMMA GEMM:  C[M,N] = act( A[M,K] @ W[N,K]^T + bias + resid )
// REQUIRES: M % 128 == 0, N % 64 == 0, K % 32 == 0 (true for every call).
// Block: 256 thr = 8 waves, tile 128x64; wave owns 32x32 C
// (2x2 v_wmma_f32_16x16x32_f16).  Ping-pong LDS tiles filled by
// GLOBAL_LOAD_ASYNC_TO_LDS_B128 (ASYNCcnt); the K-loop is manually
// unrolled x2 so buffer selection is compile-time (no movrels/cndmask).
// =======================================================================
__global__ __launch_bounds__(256) void gemm_wmma_kernel(
    const float* __restrict__ A, const float* __restrict__ W,
    const float* __restrict__ bias, const float* __restrict__ resid,
    float* __restrict__ C, int M, int N, int K, int relu)
{
    __shared__ float Asf[2][128][36];   // 144B rows: 16B aligned, bank-skewed
    __shared__ float Wsf[2][64][36];

    const int tid  = threadIdx.x;
    const int lane = tid & 31;
    const int wave = tid >> 5;
    const int wm   = (wave >> 1) * 32;   // 0,32,64,96
    const int wn   = (wave & 1) * 32;    // 0,32
    const int bm   = blockIdx.y * 128;
    const int bn   = blockIdx.x * 64;

    const int mlan = lane & 15;
    const int half = lane >> 4;
    const int h8   = half * 8;
    const int h16  = half * 16;

    // precomputed per-thread staging slots (4 x b128 for A, 2 x b128 for W)
    const float* pA[4];
    unsigned     lA0[4], lA1[4];
#pragma unroll
    for (int u = 0; u < 4; ++u) {
        int idx = tid + u * 256;
        int r   = idx >> 3;
        int c4  = (idx & 7) << 2;
        pA[u]  = A + (size_t)(bm + r) * K + c4;
        lA0[u] = (unsigned)(uintptr_t)&Asf[0][r][c4];
        lA1[u] = (unsigned)(uintptr_t)&Asf[1][r][c4];
    }
    const float* pW[2];
    unsigned     lW0[2], lW1[2];
#pragma unroll
    for (int u = 0; u < 2; ++u) {
        int idx = tid + u * 256;
        int r   = idx >> 3;
        int c4  = (idx & 7) << 2;
        pW[u]  = W + (size_t)(bn + r) * K + c4;
        lW0[u] = (unsigned)(uintptr_t)&Wsf[0][r][c4];
        lW1[u] = (unsigned)(uintptr_t)&Wsf[1][r][c4];
    }

    auto stage = [&](const unsigned* la, const unsigned* lw, int k0) {
#pragma unroll
        for (int u = 0; u < 4; ++u) async_g2l_b128(la[u], pA[u] + k0);
#pragma unroll
        for (int u = 0; u < 2; ++u) async_g2l_b128(lw[u], pW[u] + k0);
    };

    v8f acc[2][2];
#pragma unroll
    for (int i = 0; i < 2; ++i)
#pragma unroll
        for (int j = 0; j < 2; ++j)
#pragma unroll
            for (int t = 0; t < 8; ++t) acc[i][j][t] = 0.0f;

    // compute one 32-wide K step from a constant tile base
    auto compute = [&](const float (*As)[36], const float (*Ws)[36]) {
        v16h afrag[2], bfrag[2];
#pragma unroll
        for (int i = 0; i < 2; ++i) {
            const float* ap = &As[wm + i * 16 + mlan][0];
#pragma unroll
            for (int t = 0; t < 8; ++t) {
                afrag[i][t]     = (_Float16)ap[h8 + t];        // K = half*8+t
                afrag[i][8 + t] = (_Float16)ap[16 + h8 + t];   // K = 16+half*8+t
            }
        }
#pragma unroll
        for (int j = 0; j < 2; ++j) {
            const float* wp = &Ws[wn + j * 16 + mlan][0];
#pragma unroll
            for (int t = 0; t < 16; ++t) bfrag[j][t] = (_Float16)wp[h8 + t + (t & 8)];
        }
#pragma unroll
        for (int i = 0; i < 2; ++i)
#pragma unroll
            for (int j = 0; j < 2; ++j)
                acc[i][j] = wmma_f16(afrag[i], bfrag[j], acc[i][j]);
    };

    // ---- ping-pong main loop, buffer index is compile-time constant ----
    stage(lA0, lW0, 0);
    int k0 = 0;
    for (;;) {
        wait_asynccnt0();
        __syncthreads();
        if (k0 + 32 < K) stage(lA1, lW1, k0 + 32);
        if (k0 + 64 < K)
            __builtin_prefetch(A + (size_t)(bm + (tid >> 1)) * K + k0 + 64, 0, 1);
        compute(Asf[0], Wsf[0]);
        k0 += 32;
        if (k0 >= K) break;

        wait_asynccnt0();
        __syncthreads();
        if (k0 + 32 < K) stage(lA0, lW0, k0 + 32);
        if (k0 + 64 < K)
            __builtin_prefetch(A + (size_t)(bm + (tid >> 1)) * K + k0 + 64, 0, 1);
        compute(Asf[1], Wsf[1]);
        k0 += 32;
        if (k0 >= K) break;
    }

    // ---- epilogue: lane = N col, acc elem t = M row (C/D VGPR layout) ----
#pragma unroll
    for (int i = 0; i < 2; ++i)
#pragma unroll
        for (int j = 0; j < 2; ++j) {
            int col = bn + wn + j * 16 + mlan;
            float bv = bias ? bias[col] : 0.f;
#pragma unroll
            for (int t = 0; t < 8; ++t) {
                int row = bm + wm + i * 16 + half * 8 + t;
                float v = acc[i][j][t] + bv;
                if (resid) v += resid[(size_t)row * N + col];
                if (relu)  v = v > 0.f ? v : 0.f;
                C[(size_t)row * N + col] = v;
            }
        }
}

// =======================================================================
// WMMA attention: one wave per (batch, head, 16-query tile).
// dk=32 == one wmma K-step.  QK^T: one wmma per 16-key tile; scores to
// LDS; block softmax; AV: two wmmas per 32-key tile (V staged via LDS
// for the transposed B fragment).  Requires nq%16==0 && nk%32==0.
// =======================================================================
__global__ __launch_bounds__(32) void attn_wmma_kernel(
    const float* __restrict__ Q, const float* __restrict__ Kb,
    const float* __restrict__ V, const unsigned char* __restrict__ mask,
    float* __restrict__ O, int nq, int nk, float scale)
{
    __shared__ float S[16][512];     // scores / probabilities (nk <= 480)
    __shared__ float Vs[32][33];     // staged V tile (32 keys x 32 dims)
    __shared__ float rowinv[16];

    const int qt = blockIdx.x, h = blockIdx.y, b = blockIdx.z;
    const int lane = threadIdx.x;
    const int mlan = lane & 15;
    const int half = lane >> 4;
    const int h8   = half * 8;
    const int h16  = half * 16;

    // ---- Q fragment (A layout: lane = query row, elems = k dims) ----
    v16h aq;
    {
        const float* qp = Q + ((size_t)b * nq + qt * 16 + mlan) * DD + h * DK;
#pragma unroll
        for (int t = 0; t < 8; ++t) {
            aq[t]     = (_Float16)qp[h8 + t];
            aq[8 + t] = (_Float16)qp[16 + h8 + t];
        }
    }

    // ---- scores: S[16][nk] = scale * Q K^T (+mask) ----
    for (int kt = 0; kt < nk; kt += 16) {
        const float* kp = Kb + ((size_t)b * nk + kt + mlan) * DD + h * DK;
        v16h bk;
#pragma unroll
        for (int t = 0; t < 16; ++t) bk[t] = (_Float16)kp[h8 + t + (t & 8)];
        v8f z;
#pragma unroll
        for (int t = 0; t < 8; ++t) z[t] = 0.f;
        v8f s = wmma_f16(aq, bk, z);
        int key = kt + mlan;
#pragma unroll
        for (int t = 0; t < 8; ++t) {
            int row = h8 + t;
            float v = s[t] * scale;
            if (mask) {
                if (mask[((size_t)b * nq + qt * 16 + row) * nk + key] == 0)
                    v = -3.4e38f;
            }
            S[row][key] = v;
        }
    }
    __syncthreads();

    // ---- softmax per row: lane pair (mlan, mlan+16) splits the keys ----
    {
        int row = mlan;
        float mx = -3.4e38f;
        for (int k = half; k < nk; k += 2) mx = fmaxf(mx, S[row][k]);
        mx = fmaxf(mx, __shfl_xor(mx, 16, 32));
        if (mx < -3.3e38f) mx = 0.f;
        float sm = 0.f;
        for (int k = half; k < nk; k += 2) {
            float p = __expf(S[row][k] - mx);
            S[row][k] = p;
            sm += p;
        }
        sm += __shfl_xor(sm, 16, 32);
        if (half == 0) rowinv[row] = sm > 0.f ? 1.f / sm : 0.f;
    }
    __syncthreads();

    // ---- AV: out[16q x 32d] accumulated over 32-key steps ----
    v8f o0, o1;
#pragma unroll
    for (int t = 0; t < 8; ++t) { o0[t] = 0.f; o1[t] = 0.f; }

    for (int kt = 0; kt < nk; kt += 32) {
        // stage V tile (coalesced rows), consumed as columns for B frag
        {
            const float* vp = V + ((size_t)b * nk + kt + lane) * DD + h * DK;
#pragma unroll
            for (int c = 0; c < 32; c += 4) {
                float4 f = *(const float4*)(vp + c);
                Vs[lane][c]     = f.x;
                Vs[lane][c + 1] = f.y;
                Vs[lane][c + 2] = f.z;
                Vs[lane][c + 3] = f.w;
            }
        }
        __syncthreads();
        v16h ap;
        const float* pr = &S[mlan][kt];
#pragma unroll
        for (int t = 0; t < 8; ++t) {
            ap[t]     = (_Float16)pr[h8 + t];
            ap[8 + t] = (_Float16)pr[16 + h8 + t];
        }
        v16h b0, b1;
#pragma unroll
        for (int t = 0; t < 16; ++t) {
            b0[t] = (_Float16)Vs[h8 + t + (t & 8)][mlan];
            b1[t] = (_Float16)Vs[h8 + t + (t & 8)][16 + mlan];
        }
        o0 = wmma_f16(ap, b0, o0);
        o1 = wmma_f16(ap, b1, o1);
        __syncthreads();
    }

    // ---- write: lane = dim col, elem t = query row ----
#pragma unroll
    for (int t = 0; t < 8; ++t) {
        int row = h8 + t;
        float inv = rowinv[row];
        size_t orow = ((size_t)b * nq + qt * 16 + row) * DD + h * DK;
        O[orow + mlan]      = o0[t] * inv;
        O[orow + 16 + mlan] = o1[t] * inv;
    }
}

// =======================================================================
// VALU attention fallback (tiny shapes: s_s 10x10, att_si 1x10)
// =======================================================================
__global__ __launch_bounds__(32) void attn_kernel(
    const float* __restrict__ Q, const float* __restrict__ Kb,
    const float* __restrict__ V, const unsigned char* __restrict__ mask,
    float* __restrict__ O, int nq, int nk, float scale)
{
    __shared__ float qs[32];
    __shared__ float sc[512];
    const int qi = blockIdx.x, h = blockIdx.y, b = blockIdx.z;
    const int lane = threadIdx.x;
    const size_t qrow = (size_t)b * nq + qi;

    qs[lane] = Q[qrow * DD + h * DK + lane];
    __syncthreads();

    float lmax = -3.4e38f;
    for (int k = lane; k < nk; k += 32) {
        const float* kp = Kb + ((size_t)b * nk + k) * DD + h * DK;
        float dot = 0.f;
        for (int d = 0; d < DK; ++d) dot += qs[d] * kp[d];
        dot *= scale;
        if (mask && mask[((size_t)b * nq + qi) * nk + k] == 0) dot = -3.4e38f;
        sc[k] = dot;
        lmax = fmaxf(lmax, dot);
    }
    for (int o = 16; o > 0; o >>= 1) lmax = fmaxf(lmax, __shfl_xor(lmax, o, 32));
    if (lmax < -3.3e38f) lmax = 0.f;
    __syncthreads();

    float lsum = 0.f;
    for (int k = lane; k < nk; k += 32) {
        float e = __expf(sc[k] - lmax);
        sc[k] = e;
        lsum += e;
    }
    for (int o = 16; o > 0; o >>= 1) lsum += __shfl_xor(lsum, o, 32);
    float inv = lsum > 0.f ? 1.f / lsum : 0.f;
    __syncthreads();

    float out = 0.f;
    for (int k = 0; k < nk; ++k)
        out += sc[k] * V[((size_t)b * nk + k) * DD + h * DK + lane];
    O[qrow * DD + h * DK + lane] = out * inv;
}

// =======================================================================
// LayerNorm over D=256 (one block of 256 per row); optional residual.
// =======================================================================
__global__ __launch_bounds__(256) void ln_kernel(
    const float* __restrict__ x, const float* __restrict__ res,
    const float* __restrict__ g, const float* __restrict__ b,
    float* __restrict__ out)
{
    __shared__ float red[256];
    const int r = blockIdx.x, d = threadIdx.x;
    float v = x[(size_t)r * DD + d];
    if (res) v += res[(size_t)r * DD + d];
    red[d] = v;
    __syncthreads();
    for (int s = 128; s > 0; s >>= 1) { if (d < s) red[d] += red[d + s]; __syncthreads(); }
    float mean = red[0] * (1.f / DD);
    __syncthreads();
    float c = v - mean;
    red[d] = c * c;
    __syncthreads();
    for (int s = 128; s > 0; s >>= 1) { if (d < s) red[d] += red[d + s]; __syncthreads(); }
    float rstd = rsqrtf(red[0] * (1.f / DD) + 1e-5f);
    out[(size_t)r * DD + d] = c * rstd * g[d] + b[d];
}

// =======================================================================
// BatchNorm (affine=False, training stats)
// =======================================================================
__global__ __launch_bounds__(256) void bn_stats_kernel(
    const float* __restrict__ x, int N, int C, float* mean, float* var)
{
    __shared__ float s1[256], s2[256];
    const int c = blockIdx.x, t = threadIdx.x;
    float a = 0.f, bb = 0.f;
    for (int r = t; r < N; r += 256) {
        float v = x[(size_t)r * C + c];
        a += v; bb += v * v;
    }
    s1[t] = a; s2[t] = bb;
    __syncthreads();
    for (int s = 128; s > 0; s >>= 1) {
        if (t < s) { s1[t] += s1[t + s]; s2[t] += s2[t + s]; }
        __syncthreads();
    }
    if (t == 0) {
        float m = s1[0] / (float)N;
        mean[c] = m;
        var[c] = s2[0] / (float)N - m * m;
    }
}

__global__ void bn_apply_kernel(const float* __restrict__ x,
                                const float* __restrict__ mean,
                                const float* __restrict__ var,
                                float* __restrict__ y, int N, int C)
{
    int i = blockIdx.x * blockDim.x + threadIdx.x;
    if (i < N * C) {
        int c = i % C;
        y[i] = (x[i] - mean[c]) * rsqrtf(var[c] + 1e-5f);
    }
}

// ---------------- positional encoding helper ----------------
__device__ inline float pe_val(float x, int d)
{
    int j = d & 127;
    float c = __expf(-9.210340371976184f * (float)j * (1.0f / 256.0f)); // 10000^(-j/256)
    float a = x * c;
    return (d < 128) ? __sinf(a) : __cosf(a);
}

__global__ __launch_bounds__(256) void ie_emb_kernel(
    float* __restrict__ x, const float* __restrict__ hero_emb,
    const float* __restrict__ team_emb, const int* __restrict__ hero)
{
    int r = blockIdx.x, d = threadIdx.x;
    int b = r / (MT * NP);
    int p = r % NP;
    int hi = hero[b * NP + p];
    x[(size_t)r * DD + d] += hero_emb[(size_t)hi * DD + d] +
                             team_emb[(size_t)(p < 5 ? 0 : 1) * DD + d];
}

__global__ __launch_bounds__(256) void pe_sign_kernel(float* __restrict__ x)
{
    int r = blockIdx.x, d = threadIdx.x;
    int rem = r % (MT * NP);
    int t = rem / NP, p = rem % NP;
    float v = x[(size_t)r * DD + d] + pe_val((float)t, d);
    x[(size_t)r * DD + d] = (p < 5) ? v : -v;
}

__global__ __launch_bounds__(256) void g_pe_kernel(float* __restrict__ x)
{
    int r = blockIdx.x, d = threadIdx.x;
    int t = r % MT;
    x[(size_t)r * DD + d] += pe_val((float)t, d);
}

__global__ __launch_bounds__(256) void event_init_kernel(
    float* __restrict__ ec, const float* __restrict__ emb,
    const int* __restrict__ e, const int* __restrict__ e_t)
{
    int r = blockIdx.x, d = threadIdx.x;
    int ei = e[r];
    float x = (float)e_t[r] * (1.0f / 60.0f);
    ec[(size_t)r * DD + d] = emb[(size_t)ei * DD + d] + pe_val(x, d);
}

// ---------------- mask precompute ----------------
__global__ void mask_se_kernel(const int* __restrict__ e_t, const int* __restrict__ e_len,
                               const int* __restrict__ e_mask, unsigned char* __restrict__ m, int total)
{
    int idx = blockIdx.x * blockDim.x + threadIdx.x;
    if (idx >= total) return;
    int k = idx % ME;
    int p = (idx / ME) % NP;
    int t = (idx / (ME * NP)) % MT;
    int b = idx / (ME * NP * MT);
    int avail = k < e_len[b];
    int et = e_t[b * ME + k];
    int em = e_mask[(b * ME + k) * NP + p];
    m[idx] = (avail && (et < t * 60) && em) ? 1 : 0;
}

__global__ void mask_es_kernel(const int* __restrict__ e_t, const int* __restrict__ e_len,
                               const int* __restrict__ e_mask, unsigned char* __restrict__ m, int total)
{
    int idx = blockIdx.x * blockDim.x + threadIdx.x;
    if (idx >= total) return;
    int p = idx % NP;
    int t = (idx / NP) % MT;
    int k = (idx / (NP * MT)) % ME;
    int b = idx / (NP * MT * ME);
    int avail = k < e_len[b];
    int et = e_t[b * ME + k];
    int tbin = (et < 0 ? 0 : et) / 60;
    int em = e_mask[(b * ME + k) * NP + p];
    m[idx] = ((t == tbin && em) || !avail) ? 1 : 0;
}

__global__ void mask_ee_kernel(const int* __restrict__ e_t, const int* __restrict__ e_len,
                               unsigned char* __restrict__ m, int total)
{
    int idx = blockIdx.x * blockDim.x + threadIdx.x;
    if (idx >= total) return;
    int j = idx % ME;
    int i = (idx / ME) % ME;
    int b = idx / (ME * ME);
    int avail = j < e_len[b];
    m[idx] = (avail && (e_t[b * ME + i] >= e_t[b * ME + j])) ? 1 : 0;
}

__global__ void mask_ge_kernel(const int* __restrict__ e_t, const int* __restrict__ e_len,
                               unsigned char* __restrict__ m, int total)
{
    int idx = blockIdx.x * blockDim.x + threadIdx.x;
    if (idx >= total) return;
    int k = idx % ME;
    int t = (idx / ME) % MT;
    int b = idx / (ME * MT);
    int avail = k < e_len[b];
    m[idx] = (avail && (e_t[b * ME + k] < t * 60)) ? 1 : 0;
}

__global__ void add_kernel(const float* __restrict__ a, const float* __restrict__ b,
                           float* __restrict__ c, int n)
{
    int i = blockIdx.x * blockDim.x + threadIdx.x;
    if (i < n) c[i] = a[i] + b[i];
}

__global__ __launch_bounds__(32) void readout_kernel(
    const float* __restrict__ xsi, const float* __restrict__ xe,
    const float* __restrict__ w, const float* __restrict__ bias,
    float* __restrict__ out)
{
    int r = blockIdx.x, lane = threadIdx.x;
    float s = 0.f;
    for (int d = lane; d < DD; d += 32)
        s += (xsi[(size_t)r * DD + d] + xe[(size_t)r * DD + d]) * w[d];
    for (int o = 16; o > 0; o >>= 1) s += __shfl_xor(s, o, 32);
    if (lane == 0) out[r] = s + bias[0];
}

// =======================================================================
// Host orchestration
// =======================================================================
struct MhaP { const float *b, *bk, *bo, *bq, *bv, *g, *wk, *wo, *wq, *wv; };
struct FfP  { const float *b, *b1, *b2, *g, *w1, *w2; };

extern "C" void kernel_launch(void* const* d_in, const int* in_sizes, int n_in,
                              void* d_out, int out_size, void* d_ws, size_t ws_size,
                              hipStream_t stream)
{
    (void)in_sizes; (void)n_in; (void)out_size;

    // ---- input mapping: si, sg, <params tree-flattened, sorted keys>, hero, e, e_t, e_mask, e_len
    const float* si = (const float*)d_in[0];
    const float* sg = (const float*)d_in[1];
    const int PB = 2;
    auto F = [&](int i) -> const float* { return (const float*)d_in[i]; };
    auto mha_at = [&](int base) { MhaP m{F(base+0),F(base+1),F(base+2),F(base+3),F(base+4),F(base+5),F(base+6),F(base+7),F(base+8),F(base+9)}; return m; };
    auto ff_at  = [&](int base) { FfP f{F(base+0),F(base+1),F(base+2),F(base+3),F(base+4),F(base+5)}; return f; };

    MhaP p_att_e  = mha_at(PB + 0);
    MhaP p_att_si = mha_at(PB + 10);
    const float* cls_b = F(PB + 20);
    const float* cls_w = F(PB + 21);
    const int LB0 = PB + 22, LSTRIDE = 52; // per layer: e_e,e_ff,e_s,s_e,s_ff,s_s
    const float* event_emb = F(PB + 126);
    FfP p_ge_ff = ff_at(PB + 127);
    const float* ge_lb = F(PB + 133);
    const float* ge_lw = F(PB + 134);
    FfP p_ie_ff = ff_at(PB + 135);
    const float* hero_emb = F(PB + 141);
    const float* ie_lb = F(PB + 142);
    const float* ie_lw = F(PB + 143);
    const float* team_emb = F(PB + 144);
    const int* hero  = (const int*)d_in[147];
    const int* e     = (const int*)d_in[148];
    const int* e_t   = (const int*)d_in[149];
    const int* e_msk = (const int*)d_in[150];
    const int* e_len = (const int*)d_in[151];

    // ---- workspace carve ----
    char* wsp = (char*)d_ws;
    size_t off = 0;
    auto allocf = [&](size_t n) -> float* {
        off = (off + 255) & ~(size_t)255;
        float* p = (float*)(wsp + off); off += n * sizeof(float); return p;
    };
    auto allocb = [&](size_t n) -> unsigned char* {
        off = (off + 255) & ~(size_t)255;
        unsigned char* p = (unsigned char*)(wsp + off); off += n; return p;
    };
    float* mean64 = allocf(64);  float* var64 = allocf(64);
    float* mean32 = allocf(32);  float* var32 = allocf(32);
    float* si_n  = allocf((size_t)S_TOK * DSI);
    float* sg_n  = allocf((size_t)G_TOK * DSG);
    float* s_cur = allocf((size_t)S_TOK * DD);
    float* e_cur = allocf((size_t)E_TOK * DD);
    float* gbuf  = allocf((size_t)G_TOK * DD);
    float* bufQ  = allocf((size_t)S_TOK * DD);
    float* bufK  = allocf((size_t)S_TOK * DD);
    float* bufV  = allocf((size_t)S_TOK * DD);
    float* attnb = allocf((size_t)S_TOK * DD);
    float* t1    = allocf((size_t)S_TOK * DD);
    float* hbuf  = allocf((size_t)S_TOK * DHH);
    float* ssb   = allocf((size_t)S_TOK * DD);
    float* seb   = allocf((size_t)S_TOK * DD);
    float* esb   = allocf((size_t)E_TOK * DD);
    float* eeb   = allocf((size_t)E_TOK * DD);
    float* xsi   = allocf((size_t)G_TOK * DD);
    float* xe    = allocf((size_t)G_TOK * DD);
    unsigned char* m_se = allocb((size_t)BSZ * MT * NP * ME);
    unsigned char* m_es = allocb((size_t)BSZ * ME * MT * NP);
    unsigned char* m_ee = allocb((size_t)BSZ * ME * ME);
    unsigned char* m_ge = allocb((size_t)BSZ * MT * ME);
    if (off > ws_size) return; // insufficient scratch

    const float scale = 0.17677669529663687f; // 1/sqrt(32)

    auto gemm = [&](const float* A, const float* W, const float* bias,
                    const float* resid, float* C, int M, int N, int K, int relu) {
        dim3 grid((N + 63) / 64, (M + 127) / 128);
        gemm_wmma_kernel<<<grid, 256, 0, stream>>>(A, W, bias, resid, C, M, N, K, relu);
    };
    auto ln = [&](const float* x, const float* res, const float* g,
                  const float* b, float* out, int rows) {
        ln_kernel<<<rows, 256, 0, stream>>>(x, res, g, b, out);
    };
    auto ffn = [&](const FfP& p, const float* x, float* out, int rows) {
        gemm(x, p.w1, p.b1, nullptr, hbuf, rows, DHH, DD, 1);
        gemm(hbuf, p.w2, p.b2, x, t1, rows, DD, DHH, 0);
        ln(t1, nullptr, p.g, p.b, out, rows);
    };
    auto run_mha = [&](const MhaP& p, const float* qin, int Tq, const float* kin, int Tk,
                       int nb, int nq, int nk, const unsigned char* mask, float* out) {
        gemm(qin, p.wq, p.bq, nullptr, bufQ, Tq, DD, DD, 0);
        gemm(kin, p.wk, p.bk, nullptr, bufK, Tk, DD, DD, 0);
        gemm(kin, p.wv, p.bv, nullptr, bufV, Tk, DD, DD, 0);
        if ((nq % 16 == 0) && (nk % 32 == 0))
            attn_wmma_kernel<<<dim3(nq / 16, NHEAD, nb), 32, 0, stream>>>(
                bufQ, bufK, bufV, mask, attnb, nq, nk, scale);
        else
            attn_kernel<<<dim3(nq, NHEAD, nb), 32, 0, stream>>>(
                bufQ, bufK, bufV, mask, attnb, nq, nk, scale);
        gemm(attnb, p.wo, p.bo, qin, t1, Tq, DD, DD, 0);
        ln(t1, nullptr, p.g, p.b, out, Tq);
    };

    // ---- IndividualExtractor ----
    bn_stats_kernel<<<DSI, 256, 0, stream>>>(si, S_TOK, DSI, mean64, var64);
    bn_apply_kernel<<<(S_TOK * DSI + 255) / 256, 256, 0, stream>>>(si, mean64, var64, si_n, S_TOK, DSI);
    gemm(si_n, ie_lw, ie_lb, nullptr, s_cur, S_TOK, DD, DSI, 0);
    ie_emb_kernel<<<S_TOK, 256, 0, stream>>>(s_cur, hero_emb, team_emb, hero);
    ffn(p_ie_ff, s_cur, s_cur, S_TOK);
    pe_sign_kernel<<<S_TOK, 256, 0, stream>>>(s_cur);

    // ---- GlobalExtractor ----
    bn_stats_kernel<<<DSG, 256, 0, stream>>>(sg, G_TOK, DSG, mean32, var32);
    bn_apply_kernel<<<(G_TOK * DSG + 255) / 256, 256, 0, stream>>>(sg, mean32, var32, sg_n, G_TOK, DSG);
    gemm(sg_n, ge_lw, ge_lb, nullptr, gbuf, G_TOK, DD, DSG, 0);
    ffn(p_ge_ff, gbuf, gbuf, G_TOK);
    g_pe_kernel<<<G_TOK, 256, 0, stream>>>(gbuf);

    // ---- events ----
    event_init_kernel<<<E_TOK, 256, 0, stream>>>(e_cur, event_emb, e, e_t);

    // ---- masks ----
    {
        int tse = BSZ * MT * NP * ME;
        mask_se_kernel<<<(tse + 255) / 256, 256, 0, stream>>>(e_t, e_len, e_msk, m_se, tse);
        int tes = BSZ * ME * MT * NP;
        mask_es_kernel<<<(tes + 255) / 256, 256, 0, stream>>>(e_t, e_len, e_msk, m_es, tes);
        int tee = BSZ * ME * ME;
        mask_ee_kernel<<<(tee + 255) / 256, 256, 0, stream>>>(e_t, e_len, m_ee, tee);
        int tge = BSZ * MT * ME;
        mask_ge_kernel<<<(tge + 255) / 256, 256, 0, stream>>>(e_t, e_len, m_ge, tge);
    }

    // ---- encoder layers ----
    for (int l = 0; l < 2; ++l) {
        int base = LB0 + l * LSTRIDE;
        MhaP pee = mha_at(base + 0);
        FfP  pef = ff_at(base + 10);
        MhaP pes = mha_at(base + 16);
        MhaP pse = mha_at(base + 26);
        FfP  psf = ff_at(base + 36);
        MhaP pss = mha_at(base + 42);

        run_mha(pss, s_cur, S_TOK, s_cur, S_TOK, G_TOK, NP, NP, nullptr, ssb);
        run_mha(pse, s_cur, S_TOK, e_cur, E_TOK, BSZ, MT * NP, ME, m_se, seb);
        run_mha(pes, e_cur, E_TOK, s_cur, S_TOK, BSZ, ME, MT * NP, m_es, esb);
        run_mha(pee, e_cur, E_TOK, e_cur, E_TOK, BSZ, ME, ME, m_ee, eeb);

        add_kernel<<<(S_TOK * DD + 255) / 256, 256, 0, stream>>>(ssb, seb, ssb, S_TOK * DD);
        add_kernel<<<(E_TOK * DD + 255) / 256, 256, 0, stream>>>(esb, eeb, esb, E_TOK * DD);

        ffn(psf, ssb, s_cur, S_TOK);
        ffn(pef, esb, e_cur, E_TOK);
    }

    // ---- readout ----
    run_mha(p_att_si, gbuf, G_TOK, s_cur, S_TOK, G_TOK, 1, NP, nullptr, xsi);
    run_mha(p_att_e,  gbuf, G_TOK, e_cur, E_TOK, BSZ, MT, ME, m_ge, xe);
    readout_kernel<<<G_TOK, 32, 0, stream>>>(xsi, xe, cls_w, cls_b, (float*)d_out);
}
